// WindowAttention_69939247448615
// MI455X (gfx1250) — compile-verified, matrix operand order checked by reading, not run on hardware
//
#include <hip/hip_runtime.h>
#include <hip/hip_bf16.h>

typedef __attribute__((ext_vector_type(4)))  _Float16 v4h;
typedef __attribute__((ext_vector_type(8)))  _Float16 v8h;
typedef __attribute__((ext_vector_type(16))) _Float16 v16h;
typedef __attribute__((ext_vector_type(8)))  float    v8f;

#define N_TOK   49
#define NPAD    64
#define DIM     512
#define HEADS   16
#define HD      32
#define NW      64
#define QKV_DIM 1536

// ---- LDS layout (bytes) ----
#define OFF_XS    0                  // x staging [64][512] f16; later per-wave P buffers (8KB/wave)
#define OFF_Q     65536              // q [64][512] f16; later O buffer
#define OFF_K     131072             // k [64][512] f16
#define OFF_VT    196608             // v^T [512][64] f16
#define OFF_MASK  262144             // mask slice [49*49] f32
#define OFF_RELB  271748             // bias_table [169*16] f32
#define OFF_RELI  282564             // rel_idx [49*49] i32
#define SMEM_BYTES 292168
static_assert(SMEM_BYTES <= 320 * 1024, "LDS budget");

// Load a 16x32 f16 A-style fragment (rows = M or N dim, K contiguous) from a
// row-major [rows][stride] array. Per wave32-lane layout (CDNA5 ISA 7.12.2):
//   lanes 0-15 : row = row0+lane, elems 0..7 -> K=k0..k0+7,  elems 8..15 -> K=k0+16..k0+23
//   lanes16-31 : row = row0+lane-16, elems 0..7 -> K=k0+8..15, elems 8..15 -> K=k0+24..31
__device__ __forceinline__ v16h load_frag(const _Float16* base, int row0, int stride, int k0) {
  const int l  = threadIdx.x & 31;
  const int hs = l >> 4;
  const _Float16* p = base + (row0 + (l & 15)) * stride + k0 + hs * 8;
  v8h lo = *(const v8h*)(p);
  v8h hi = *(const v8h*)(p + 16);
  return __builtin_shufflevector(lo, hi, 0,1,2,3,4,5,6,7,8,9,10,11,12,13,14,15);
}

__device__ __forceinline__ v8f wmma_f16(v16h a, v16h b, v8f c) {
  // D = A x B + C, f32 accumulate
  return __builtin_amdgcn_wmma_f32_16x16x32_f16(false, a, false, b, (short)0, c, false, false);
}

// fp32 -> f16 weight conversion (runs once per launch; weights then live in L2)
__global__ void cvt_weights_kernel(const float* __restrict__ qkv_w,
                                   const float* __restrict__ proj_w,
                                   _Float16* __restrict__ wq,
                                   _Float16* __restrict__ wp) {
  const int i = blockIdx.x * blockDim.x + threadIdx.x;
  if (i < QKV_DIM * DIM) wq[i] = (_Float16)qkv_w[i];
  if (i < DIM * DIM)     wp[i] = (_Float16)proj_w[i];
}

__global__ __launch_bounds__(256, 1)
void win_attn_fused(const float* __restrict__ x,
                    const float* __restrict__ mask,
                    const _Float16* __restrict__ wq,
                    const float* __restrict__ qkv_b,
                    const _Float16* __restrict__ wp,
                    const float* __restrict__ proj_b,
                    const float* __restrict__ bias_table,
                    const int* __restrict__ rel_idx,
                    float* __restrict__ out) {
  extern __shared__ char smem[];
  _Float16* xs     = (_Float16*)(smem + OFF_XS);
  _Float16* qbuf   = (_Float16*)(smem + OFF_Q);
  _Float16* kbuf   = (_Float16*)(smem + OFF_K);
  _Float16* vT     = (_Float16*)(smem + OFF_VT);
  float*    mask_s = (float*)(smem + OFF_MASK);
  float*    relb_s = (float*)(smem + OFF_RELB);
  int*      reli_s = (int*)  (smem + OFF_RELI);

  const int b    = blockIdx.x;
  const int tid  = threadIdx.x;
  const int wave = tid >> 5;
  const int lane = tid & 31;
  const int nlo  = lane & 15;
  const int hsp  = (lane >> 4) * 8;      // row offset for C-fragment halves
  const v8f vzero = {0.f, 0.f, 0.f, 0.f, 0.f, 0.f, 0.f, 0.f};

  // ---------------- Phase 1: stage x (f32 -> f16), mask, bias tables ----------------
  {
    const float* xb = x + (size_t)b * N_TOK * DIM;
    for (int i = tid; i < NPAD * (DIM / 4); i += 256) {
      const int row = i >> 7;            // 128 float4 groups per row
      const int c4  = (i & 127) << 2;
      v4h hv;
      if (row < N_TOK) {
        const float4 f = *(const float4*)(xb + row * DIM + c4);
        hv[0] = (_Float16)f.x; hv[1] = (_Float16)f.y;
        hv[2] = (_Float16)f.z; hv[3] = (_Float16)f.w;
      } else {
        hv[0] = (_Float16)0.f; hv[1] = (_Float16)0.f;
        hv[2] = (_Float16)0.f; hv[3] = (_Float16)0.f;
      }
      *(v4h*)(xs + row * DIM + c4) = hv;
    }
    const int g = b & (NW - 1);          // mask group = window index within image
    for (int i = tid; i < N_TOK * N_TOK; i += 256) {
      mask_s[i] = mask[(size_t)g * N_TOK * N_TOK + i];
      reli_s[i] = rel_idx[i];
    }
    for (int i = tid; i < 169 * HEADS; i += 256) relb_s[i] = bias_table[i];
  }
  __syncthreads();

  // ---------------- Phase 2: QKV GEMM  [64 x 1536] = xs[64x512] @ wq^T ----------------
  // Two N-tiles per pass: each A fragment feeds 2 WMMAs (halves LDS read traffic).
  const float qscale = 0.17677669529663689f;   // HD^-0.5

  auto store_qkv = [&](int jt, const v8f* accv) {
    const int   ncol = jt * 16 + nlo;          // column of qkv output
    const int   s    = ncol >> 9;              // 0=q 1=k 2=v
    const int   c    = ncol & 511;
    const float qb   = qkv_b[ncol];
    #pragma unroll
    for (int mt = 0; mt < 4; ++mt) {
      #pragma unroll
      for (int r = 0; r < 8; ++r) {
        const int m = mt * 16 + r + hsp;
        const float val = accv[mt][r] + qb;
        if (s == 0)      qbuf[m * DIM + c] = (_Float16)(val * qscale);
        else if (s == 1) kbuf[m * DIM + c] = (_Float16)val;
        else             vT[c * NPAD + m]  = (_Float16)val;   // store v transposed
      }
    }
  };

  for (int p = 0; p < 6; ++p) {                // 96 tiles / 8 waves / 2-per-pass
    const int jt0 = wave + p * 16;
    const int jt1 = jt0 + 8;
    if (p < 5) {                               // prefetch next pair's weight rows
      const int njt0 = jt0 + 16, njt1 = jt1 + 16;
      const int prow = (lane < 16) ? (njt0 * 16 + lane) : (njt1 * 16 + (lane - 16));
      __builtin_prefetch(wq + (size_t)prow * DIM, 0, 0);     // global_prefetch_b8
    }
    v8f acc[2][4];
    #pragma unroll
    for (int t = 0; t < 2; ++t)
      #pragma unroll
      for (int mt = 0; mt < 4; ++mt) acc[t][mt] = vzero;
    for (int kt = 0; kt < DIM / 32; ++kt) {
      const v16h b0 = load_frag(wq, jt0 * 16, DIM, kt * 32);
      const v16h b1 = load_frag(wq, jt1 * 16, DIM, kt * 32);
      #pragma unroll
      for (int mt = 0; mt < 4; ++mt) {
        const v16h af = load_frag(xs, mt * 16, DIM, kt * 32);
        acc[0][mt] = wmma_f16(af, b0, acc[0][mt]);
        acc[1][mt] = wmma_f16(af, b1, acc[1][mt]);
      }
    }
    store_qkv(jt0, acc[0]);
    store_qkv(jt1, acc[1]);
  }
  __syncthreads();

  // ---------------- Phase 3: attention, 2 heads per wave ----------------
  v8f o[2][4][2];
  #pragma unroll
  for (int hh = 0; hh < 2; ++hh)
    #pragma unroll
    for (int mt = 0; mt < 4; ++mt)
      #pragma unroll
      for (int dt = 0; dt < 2; ++dt) o[hh][mt][dt] = vzero;

  _Float16* pb = (_Float16*)(smem + OFF_XS + wave * 8192);   // per-wave P [64][64] f16

  for (int hh = 0; hh < 2; ++hh) {
    const int h = wave * 2 + hh;
    const _Float16* qh = qbuf + h * HD;
    const _Float16* kh = kbuf + h * HD;

    v16h bk[4];
    #pragma unroll
    for (int nt = 0; nt < 4; ++nt) bk[nt] = load_frag(kh, nt * 16, DIM, 0);

    for (int mt = 0; mt < 4; ++mt) {
      const v16h aq = load_frag(qh, mt * 16, DIM, 0);
      v8f sv[4];
      #pragma unroll
      for (int nt = 0; nt < 4; ++nt) sv[nt] = wmma_f16(aq, bk[nt], vzero);   // K=32 in one op

      // relative-position bias + shifted-window mask + -inf padding (C layout)
      #pragma unroll
      for (int nt = 0; nt < 4; ++nt) {
        const int n = nt * 16 + nlo;
        #pragma unroll
        for (int r = 0; r < 8; ++r) {
          const int m = mt * 16 + r + hsp;
          float t = sv[nt][r];
          if (n < N_TOK && m < N_TOK)
            t += relb_s[reli_s[m * N_TOK + n] * HEADS + h] + mask_s[m * N_TOK + n];
          if (n >= N_TOK) t = -1e30f;
          sv[nt][r] = t;
        }
      }
      // row softmax: reduce across 4 N-tiles + 16 lanes (both halves independently)
      #pragma unroll
      for (int r = 0; r < 8; ++r) {
        float mx = fmaxf(fmaxf(sv[0][r], sv[1][r]), fmaxf(sv[2][r], sv[3][r]));
        #pragma unroll
        for (int off = 1; off < 16; off <<= 1) mx = fmaxf(mx, __shfl_xor(mx, off, 16));
        float sum = 0.f;
        #pragma unroll
        for (int nt = 0; nt < 4; ++nt) {
          const float e = __expf(sv[nt][r] - mx);
          sv[nt][r] = e; sum += e;
        }
        #pragma unroll
        for (int off = 1; off < 16; off <<= 1) sum += __shfl_xor(sum, off, 16);
        const float inv = 1.0f / sum;
        const int m = mt * 16 + r + hsp;
        #pragma unroll
        for (int nt = 0; nt < 4; ++nt)
          pb[m * NPAD + nt * 16 + nlo] = (_Float16)(sv[nt][r] * inv);   // C->A layout via LDS
      }
    }

    // O = P @ V   (B operand from v^T: rows = head-dim, K = token contiguous)
    const _Float16* vTh = vT + (h * HD) * NPAD;
    #pragma unroll
    for (int kt = 0; kt < 2; ++kt) {
      const v16h bv0 = load_frag(vTh, 0,  NPAD, kt * 32);
      const v16h bv1 = load_frag(vTh, 16, NPAD, kt * 32);
      #pragma unroll
      for (int mt = 0; mt < 4; ++mt) {
        const v16h ap = load_frag(pb, mt * 16, NPAD, kt * 32);
        o[hh][mt][0] = wmma_f16(ap, bv0, o[hh][mt][0]);
        o[hh][mt][1] = wmma_f16(ap, bv1, o[hh][mt][1]);
      }
    }
  }
  __syncthreads();   // all waves done reading q/k/vT

  // ---------------- Phase 3b: spill O registers into LDS (reuse q region) ----------------
  _Float16* obuf = qbuf;
  #pragma unroll
  for (int hh = 0; hh < 2; ++hh)
    #pragma unroll
    for (int mt = 0; mt < 4; ++mt)
      #pragma unroll
      for (int dt = 0; dt < 2; ++dt) {
        const int col = (wave * 2 + hh) * HD + dt * 16 + nlo;
        #pragma unroll
        for (int r = 0; r < 8; ++r) {
          const int m = mt * 16 + r + hsp;
          obuf[m * DIM + col] = (_Float16)o[hh][mt][dt][r];
        }
      }
  __syncthreads();

  // ---------------- Phase 4: proj GEMM  out = O @ wp^T + proj_b ----------------
  float* outb = out + (size_t)b * N_TOK * DIM;

  auto store_out = [&](int jt, const v8f* accv) {
    const int   j   = jt * 16 + nlo;
    const float pbv = proj_b[j];
    #pragma unroll
    for (int mt = 0; mt < 4; ++mt)
      #pragma unroll
      for (int r = 0; r < 8; ++r) {
        const int m = mt * 16 + r + hsp;
        if (m < N_TOK) outb[m * DIM + j] = accv[mt][r] + pbv;
      }
  };

  #pragma unroll
  for (int p = 0; p < 2; ++p) {                // 32 tiles / 8 waves / 2-per-pass
    const int jt0 = wave + p * 16;
    const int jt1 = jt0 + 8;
    v8f acc[2][4];
    #pragma unroll
    for (int t = 0; t < 2; ++t)
      #pragma unroll
      for (int mt = 0; mt < 4; ++mt) acc[t][mt] = vzero;
    for (int kt = 0; kt < DIM / 32; ++kt) {
      const v16h b0 = load_frag(wp, jt0 * 16, DIM, kt * 32);
      const v16h b1 = load_frag(wp, jt1 * 16, DIM, kt * 32);
      #pragma unroll
      for (int mt = 0; mt < 4; ++mt) {
        const v16h af = load_frag(obuf, mt * 16, DIM, kt * 32);
        acc[0][mt] = wmma_f16(af, b0, acc[0][mt]);
        acc[1][mt] = wmma_f16(af, b1, acc[1][mt]);
      }
    }
    store_out(jt0, acc[0]);
    store_out(jt1, acc[1]);
  }
}

extern "C" void kernel_launch(void* const* d_in, const int* in_sizes, int n_in,
                              void* d_out, int out_size, void* d_ws, size_t ws_size,
                              hipStream_t stream) {
  const float* x       = (const float*)d_in[0];
  const float* mask    = (const float*)d_in[1];
  const float* qkv_w   = (const float*)d_in[2];
  const float* qkv_b   = (const float*)d_in[3];
  const float* proj_w  = (const float*)d_in[4];
  const float* proj_b  = (const float*)d_in[5];
  const float* bias_t  = (const float*)d_in[6];
  const int*   rel_idx = (const int*)d_in[7];
  float* out = (float*)d_out;

  _Float16* wq = (_Float16*)d_ws;                       // [1536][512] f16
  _Float16* wp = wq + (size_t)QKV_DIM * DIM;            // [512][512]  f16

  cvt_weights_kernel<<<(QKV_DIM * DIM + 255) / 256, 256, 0, stream>>>(qkv_w, proj_w, wq, wp);

  (void)hipFuncSetAttribute(reinterpret_cast<const void*>(win_attn_fused),
                            hipFuncAttributeMaxDynamicSharedMemorySize, SMEM_BYTES);
  win_attn_fused<<<2048, 256, SMEM_BYTES, stream>>>(x, mask, wq, qkv_b, wp, proj_b,
                                                    bias_t, rel_idx, out);
}
